// GraphAttentionLayer_82248623718465
// MI455X (gfx1250) — compile-verified, matrix operand order checked by reading, not run on hardware
//
#include <hip/hip_runtime.h>
#include <hip/hip_bf16.h>
#include <stdint.h>

#define N_NODES 4096
#define IN_F    512
#define OUT_F   128
#define HEADS   8
#define HF      (HEADS * OUT_F)   // 1024
#define GAT_ALPHA 0.2f
#define MASK_VAL (-9.0e15f)
#define LOG2E   1.44269504088896f

typedef __attribute__((ext_vector_type(16))) __bf16 v16bf;
typedef __attribute__((ext_vector_type(8)))  float  v8f;

union Frag16 {
    unsigned short us[16];
    uint4          u4[2];
    v16bf          v;
};
union I4x2 { int4   v[2]; int   a[8]; };
union F4x2 { float4 v[2]; float f[8]; };

__device__ __forceinline__ unsigned short f2bf(float f) {
    union { float f; unsigned int u; } x; x.f = f;
    unsigned int u = x.u;
    unsigned int r = u + 0x7FFFu + ((u >> 16) & 1u);   // RNE
    return (unsigned short)(r >> 16);
}

// ---------------------------------------------------------------- converts
__global__ __launch_bounds__(256) void k_conv_x(const float* __restrict__ X,
                                                unsigned short* __restrict__ Xb) {
    int t = blockIdx.x * 256 + threadIdx.x;
    if (t < N_NODES * IN_F) Xb[t] = f2bf(X[t]);
}

__global__ __launch_bounds__(256) void k_conv_wT(const float* __restrict__ W,
                                                 unsigned short* __restrict__ WbT) {
    int t = blockIdx.x * 256 + threadIdx.x;      // over IN_F*HF
    if (t < IN_F * HF) {
        int k = t / HF, n = t % HF;
        WbT[(size_t)n * IN_F + k] = f2bf(W[t]);  // WbT[n][k] = W[k][n]
    }
}

// ------------------------------------------- projection GEMM: h = Xb @ W
// one wave per 16x16 tile; A rows lane-contiguous in K, B from WbT rows.
__global__ __launch_bounds__(256) void k_gemm_proj(
    const unsigned short* __restrict__ Xb,    // N x IN_F   bf16
    const unsigned short* __restrict__ WbT,   // HF x IN_F  bf16
    float* __restrict__ hf,                   // N x HF     f32
    unsigned short* __restrict__ hbT)         // HF x N     bf16 (transposed)
{
    int lane = threadIdx.x & 31;
    int wid  = threadIdx.x >> 5;
    int tile = blockIdx.x * 8 + wid;          // 256*64 = 16384 tiles
    int ti = tile >> 6;                       // i-tile 0..255
    int tf = tile & 63;                       // f-tile 0..63
    int i0 = ti * 16, f0 = tf * 16;
    int mn    = lane & 15;
    int kbase = (lane >> 4) * 8;

    const unsigned short* arow = Xb  + (size_t)(i0 + mn) * IN_F;
    const unsigned short* brow = WbT + (size_t)(f0 + mn) * IN_F;

    v8f c = {};
    #pragma unroll
    for (int k0 = 0; k0 < IN_F; k0 += 32) {
        Frag16 fa, fb;
        const uint4* pa = (const uint4*)(arow + k0 + kbase);
        fa.u4[0] = pa[0]; fa.u4[1] = pa[2];   // K = k0+kb+0..7 , k0+kb+16..23
        const uint4* pb = (const uint4*)(brow + k0 + kbase);
        fb.u4[0] = pb[0]; fb.u4[1] = pb[2];
        c = __builtin_amdgcn_wmma_f32_16x16x32_bf16(false, fa.v, false, fb.v,
                                                    (short)0, c, false, false);
    }
    int moff = (lane >> 4) * 8;
    int n    = lane & 15;
    #pragma unroll
    for (int v = 0; v < 8; ++v) {
        float val = c[v];
        int gi = i0 + moff + v;
        int gf = f0 + n;
        hf[(size_t)gi * HF + gf]       = val;
        hbT[(size_t)gf * N_NODES + gi] = f2bf(val);
    }
}

// ------------------------------------------------- left/right projections
// also emits rightT[hd][j] pre-scaled by log2(e) for the aggregation kernel
__global__ __launch_bounds__(256) void k_left_right(const float* __restrict__ hf,
                                                    const float* __restrict__ a,
                                                    float* __restrict__ left,
                                                    float* __restrict__ right,
                                                    float* __restrict__ rightT) {
    int t = blockIdx.x * 256 + threadIdx.x;   // N*HEADS
    if (t >= N_NODES * HEADS) return;
    int i = t >> 3, hd = t & 7;
    const float* hrow = hf + (size_t)i * HF + hd * OUT_F;
    float l = 0.f, r = 0.f;
    #pragma unroll 4
    for (int f = 0; f < OUT_F; ++f) {
        float hv = hrow[f];
        l += hv * a[f * HEADS + hd];
        r += hv * a[(OUT_F + f) * HEADS + hd];
    }
    left[t] = l; right[t] = r;
    rightT[(size_t)hd * N_NODES + i] = r * LOG2E;
}

// ------------------------------------- per-(i,head) softmax max & denominator
__global__ __launch_bounds__(256) void k_softmax_stats(
    const int*   __restrict__ adj,
    const float* __restrict__ left,
    const float* __restrict__ right,
    float* __restrict__ rowmax,
    float* __restrict__ rowsum)
{
    int lane = threadIdx.x & 31;
    int wid  = threadIdx.x >> 5;
    int i = blockIdx.x * 8 + wid;
    const int* arow = adj + (size_t)i * N_NODES;

    float lf[HEADS], m[HEADS], s[HEADS];
    #pragma unroll
    for (int hd = 0; hd < HEADS; ++hd) {
        lf[hd] = left[i * HEADS + hd];
        m[hd] = MASK_VAL; s[hd] = 0.f;
    }
    for (int j = lane; j < N_NODES; j += 32) {
        int av = arow[j];
        const float* rr = right + (size_t)j * HEADS;
        #pragma unroll
        for (int hd = 0; hd < HEADS; ++hd) {
            float e;
            if (av > 0) { float x = lf[hd] + rr[hd]; e = x > 0.f ? x : GAT_ALPHA * x; }
            else          e = MASK_VAL;
            float nm = fmaxf(m[hd], e);
            s[hd] = s[hd] * __expf(m[hd] - nm) + __expf(e - nm);
            m[hd] = nm;
        }
    }
    #pragma unroll
    for (int hd = 0; hd < HEADS; ++hd) {
        float mm = m[hd], ss = s[hd];
        #pragma unroll
        for (int off = 16; off > 0; off >>= 1) {
            float om = __shfl_xor(mm, off, 32);
            float os = __shfl_xor(ss, off, 32);
            float nm = fmaxf(mm, om);
            ss = ss * __expf(mm - nm) + os * __expf(om - nm);
            mm = nm;
        }
        if (lane == 0) { rowmax[i * HEADS + hd] = mm; rowsum[i * HEADS + hd] = ss; }
    }
}

// ----------------------- attention aggregation: out = softmax(e) @ h, + ELU
// block = 8 waves (one per head), one 16-row i-tile per block.
// probabilities kept unnormalized in the K-loop; 1/rowsum applied per C-row
// in the epilogue (WMMA is linear in A).
__global__ __launch_bounds__(256) void k_attn_agg(
    const int*   __restrict__ adj,
    const float* __restrict__ left,
    const float* __restrict__ rightT,         // HEADS x N, pre-scaled by log2e
    const float* __restrict__ rowmax,
    const float* __restrict__ rowsum,
    const unsigned short* __restrict__ hbT,   // HF x N bf16
    float* __restrict__ out)                  // N x HF f32
{
    int lane = threadIdx.x & 31;
    int hd   = threadIdx.x >> 5;              // head = wave id
    int i0   = blockIdx.x * 16;
    int mA    = lane & 15;
    int kbase = (lane >> 4) * 8;
    int iA = i0 + mA;

    float lf2 = left  [iA * HEADS + hd] * LOG2E;
    float mx2 = rowmax[iA * HEADS + hd] * LOG2E;
    const int*   arow   = adj    + (size_t)iA * N_NODES;
    const float* rt_row = rightT + (size_t)hd * N_NODES;

    const unsigned short* brow[8];
    #pragma unroll
    for (int t = 0; t < 8; ++t)
        brow[t] = hbT + (size_t)(hd * OUT_F + t * 16 + mA) * N_NODES;

    v8f c[8];
    #pragma unroll
    for (int t = 0; t < 8; ++t) c[t] = (v8f){};

    for (int j0 = 0; j0 < N_NODES; j0 += 32) {
        __builtin_prefetch(arow + j0 + 128, 0, 1);    // global_prefetch_b8
        // adj + rightT for the two contiguous 8-element K runs of this lane
        I4x2 adjU[2]; F4x2 rtU[2];
        const int4* ap = (const int4*)(arow + j0 + kbase);
        adjU[0].v[0] = ap[0]; adjU[0].v[1] = ap[1];   // k = kbase+0..7
        adjU[1].v[0] = ap[4]; adjU[1].v[1] = ap[5];   // k = kbase+16..23
        const float4* rp = (const float4*)(rt_row + j0 + kbase);
        rtU[0].v[0] = rp[0]; rtU[0].v[1] = rp[1];
        rtU[1].v[0] = rp[4]; rtU[1].v[1] = rp[5];

        Frag16 fa;
        #pragma unroll
        for (int p = 0; p < 16; ++p) {
            int g = p >> 3, idx = p & 7;
            float x = lf2 + rtU[g].f[idx];            // log2e * (l + r)
            x = x > 0.f ? x : GAT_ALPHA * x;          // leaky (pos. homogeneous)
            float pr = exp2f(x - mx2);                // bare v_exp_f32
            pr = (adjU[g].a[idx] > 0) ? pr : 0.f;
            fa.us[p] = f2bf(pr);
        }
        #pragma unroll
        for (int t = 0; t < 8; ++t) {
            Frag16 fb;
            const uint4* pb = (const uint4*)(brow[t] + j0 + kbase);
            fb.u4[0] = pb[0]; fb.u4[1] = pb[2];
            c[t] = __builtin_amdgcn_wmma_f32_16x16x32_bf16(false, fa.v, false, fb.v,
                                                           (short)0, c[t], false, false);
        }
    }
    int moff = (lane >> 4) * 8;
    int nC   = lane & 15;
    float invv[8];
    #pragma unroll
    for (int v = 0; v < 8; ++v)
        invv[v] = 1.0f / rowsum[(i0 + moff + v) * HEADS + hd];
    #pragma unroll
    for (int t = 0; t < 8; ++t) {
        #pragma unroll
        for (int v = 0; v < 8; ++v) {
            float x = c[t][v] * invv[v];              // softmax normalization
            float y = x > 0.f ? x : expm1f(x);        // ELU
            int gi = i0 + moff + v;
            int gf = hd * OUT_F + t * 16 + nC;
            out[(size_t)gi * HF + gf] = y;
        }
    }
}

// ---------------------------------------------------------------- launcher
extern "C" void kernel_launch(void* const* d_in, const int* in_sizes, int n_in,
                              void* d_out, int out_size, void* d_ws, size_t ws_size,
                              hipStream_t stream) {
    const float* X   = (const float*)d_in[0];   // 4096 x 512
    const int*   adj = (const int*)  d_in[1];   // 4096 x 4096
    const float* W   = (const float*)d_in[2];   // 512 x 1024
    const float* a   = (const float*)d_in[3];   // 256 x 8
    float* out = (float*)d_out;                 // 4096 x 1024

    char* ws = (char*)d_ws;
    unsigned short* hbT = (unsigned short*)(ws);                     // 8 MB
    float*          hf  = (float*)         (ws + (8  << 20));        // 16 MB
    unsigned short* Xb  = (unsigned short*)(ws + (24 << 20));        // 4 MB
    unsigned short* WbT = (unsigned short*)(ws + (28 << 20));        // 1 MB
    float* left   = (float*)(ws + (29 << 20));                       // 128 KB each
    float* right  = left   + N_NODES * HEADS;
    float* rowmax = right  + N_NODES * HEADS;
    float* rowsum = rowmax + N_NODES * HEADS;
    float* rightT = rowsum + N_NODES * HEADS;                        // 8 x 4096

    k_conv_x   <<<(N_NODES * IN_F) / 256, 256, 0, stream>>>(X, Xb);
    k_conv_wT  <<<(IN_F * HF) / 256,      256, 0, stream>>>(W, WbT);
    k_gemm_proj<<<(N_NODES / 16) * (HF / 16) / 8, 256, 0, stream>>>(Xb, WbT, hf, hbT);
    k_left_right<<<(N_NODES * HEADS) / 256, 256, 0, stream>>>(hf, a, left, right, rightT);
    k_softmax_stats<<<N_NODES / 8, 256, 0, stream>>>(adj, left, right, rowmax, rowsum);
    k_attn_agg<<<N_NODES / 16, 256, 0, stream>>>(adj, left, rightT, rowmax, rowsum, hbT, out);
}
